// NMT_65515431133654
// MI455X (gfx1250) — matrix operand order repaired and checked by reading, not compiled, not run on hardware
//
#include <hip/hip_runtime.h>
#include <hip/hip_bf16.h>

// ---------------------------------------------------------------------------
// Problem constants (from the reference)
// ---------------------------------------------------------------------------
#define BB 32
#define SS 128
#define EE 256
#define UU 256
#define VV 32000
#define G3 768  /* 3*U */

typedef __attribute__((ext_vector_type(16))) _Float16 v16h;
typedef __attribute__((ext_vector_type(8)))  _Float16 v8h;
typedef __attribute__((ext_vector_type(8)))  float    v8f;

// ---------------------------------------------------------------------------
// WMMA fragment helpers (16x16x32 f16 -> f32), wave32.
// A: [16,32] f16 row-major source (lda elems). Lane L: m=L&15, g=L>>4.
//    element e: k = e + 8*g            (e<8)
//               k = (e-8) + 16 + 8*g   (e>=8)
//    -> two contiguous 8-half (16B) loads at kbase+8g and kbase+8g+16.
// B: Bt stored N-major [N,K]. Lane L: n=L&15, g=L>>4; element e: same k as A
//    -> two contiguous 8-half loads at kbase+8g, kbase+8g+16.
// C/D: lane L: n=L&15; row r -> m = r + 8*(L>>4).
// ---------------------------------------------------------------------------
__device__ __forceinline__ v16h load_a16(const _Float16* A, int lda, int kbase) {
  int lane = threadIdx.x & 31;
  int g = lane >> 4, m = lane & 15;
  const _Float16* p = A + m * lda + kbase + 8 * g;
  v8h lo = *(const v8h*)(p);
  v8h hi = *(const v8h*)(p + 16);
  v16h a;
#pragma unroll
  for (int i = 0; i < 8; ++i) { a[i] = lo[i]; a[8 + i] = hi[i]; }
  return a;
}

__device__ __forceinline__ v16h load_b16(const _Float16* Bt, int ldb, int kbase) {
  int lane = threadIdx.x & 31;
  int g = lane >> 4, n = lane & 15;
  const _Float16* p = Bt + n * ldb + kbase + 8 * g;
  v8h lo = *(const v8h*)(p);
  v8h hi = *(const v8h*)(p + 16);
  v16h b;
#pragma unroll
  for (int i = 0; i < 8; ++i) { b[i] = lo[i]; b[8 + i] = hi[i]; }
  return b;
}

__device__ __forceinline__ v8f wmma16(const v16h& a, const v16h& b, const v8f& c) {
  return __builtin_amdgcn_wmma_f32_16x16x32_f16(false, a, false, b,
                                                (short)0, c, false, false);
}

// Accumulate a full K-strip into a single 16x16 f32 tile (used by scan kernels).
__device__ __forceinline__ v8f wmma_strip(const _Float16* A, int lda,
                                          const _Float16* Bt, int ldb,
                                          int K, v8f acc) {
#pragma unroll 4
  for (int k = 0; k < K; k += 32) {
    v16h a = load_a16(A, lda, k);
    v16h b = load_b16(Bt, ldb, k);
    acc = wmma16(a, b, acc);
  }
  return acc;
}

__device__ __forceinline__ void store_tile(float* C, int ldc, const v8f& d,
                                           const float* bias /*nullable, +n0*/) {
  int lane = threadIdx.x & 31;
  int g = lane >> 4, n = lane & 15;
#pragma unroll
  for (int r = 0; r < 8; ++r) {
    float v = d[r];
    if (bias) v += bias[n];
    C[(r + 8 * g) * ldc + n] = v;
  }
}

__device__ __forceinline__ float sigf(float x) { return 1.f / (1.f + __expf(-x)); }

// ---------------------------------------------------------------------------
// Generic WMMA GEMM, 2x2-tile blocked: each wave produces a 32x32 output
// block (4 WMMA tiles) so every k-step does 4 fragment loads -> 4 WMMAs
// (1:1 wmma:load ratio; A and B fragments each reused twice).
// C[M,N] = A[M,K](f16 row-major) * Bt[N,K](f16, N-major) + bias.
// Requires M,N multiples of 32 and K multiple of 32 (true for all uses).
// ---------------------------------------------------------------------------
__global__ __launch_bounds__(256) void gemm_f16_wmma(
    const _Float16* __restrict__ A, const _Float16* __restrict__ Bt,
    const float* __restrict__ bias, float* __restrict__ C,
    int ldc, int M, int N, int K) {
  int wave = blockIdx.x * (blockDim.x >> 5) + (threadIdx.x >> 5);
  int mb = M >> 5;
  int jobs = mb * (N >> 5);
  if (wave >= jobs) return;            // wave-uniform: EXEC all-ones inside
  int mi = wave % mb, ni = wave / mb;

  const _Float16* A0 = A + (size_t)(mi * 32) * K;
  const _Float16* A1 = A0 + 16 * K;
  const _Float16* B0 = Bt + (size_t)(ni * 32) * K;
  const _Float16* B1 = B0 + 16 * K;

  v8f acc00 = {}, acc01 = {}, acc10 = {}, acc11 = {};
#pragma unroll 2
  for (int k = 0; k < K; k += 32) {
    v16h a0 = load_a16(A0, K, k);
    v16h a1 = load_a16(A1, K, k);
    v16h b0 = load_b16(B0, K, k);
    v16h b1 = load_b16(B1, K, k);
    acc00 = wmma16(a0, b0, acc00);
    acc01 = wmma16(a0, b1, acc01);
    acc10 = wmma16(a1, b0, acc10);
    acc11 = wmma16(a1, b1, acc11);
  }
  float* C00 = C + (size_t)(mi * 32) * ldc + ni * 32;
  const float* bs0 = bias ? bias + ni * 32 : nullptr;
  const float* bs1 = bias ? bias + ni * 32 + 16 : nullptr;
  store_tile(C00,                ldc, acc00, bs0);
  store_tile(C00 + 16,           ldc, acc01, bs1);
  store_tile(C00 + 16 * ldc,     ldc, acc10, bs0);
  store_tile(C00 + 16 * ldc + 16, ldc, acc11, bs1);
}

// ---------------------------------------------------------------------------
// Embedding gather + f32->f16 convert. shift=1: decoder teacher forcing
// (token[b,0]=SOS=0, token[b,s]=labels[b,s-1]).
// ---------------------------------------------------------------------------
__global__ void gather_embed_f16(const int* __restrict__ toks,
                                 const int* __restrict__ labels,
                                 const float* __restrict__ embed,
                                 _Float16* __restrict__ out,
                                 int BS, int E, int S, int shift) {
  int stride = gridDim.x * blockDim.x;
  for (int i = blockIdx.x * blockDim.x + threadIdx.x; i < BS * E; i += stride) {
    int row = i / E, e = i - row * E;
    int tok;
    if (shift) { int s = row % S; tok = (s == 0) ? 0 : labels[row - 1]; }
    else        tok = toks[row];
    out[i] = (_Float16)embed[(size_t)tok * E + e];
  }
}

// f32 [rowOff+K, N] slice -> f16 transposed [N, K]
__global__ void convT_f16(const float* __restrict__ in, _Float16* __restrict__ out,
                          int N, int K, int rowOff) {
  int stride = gridDim.x * blockDim.x;
  int total = N * K;
  for (int i = blockIdx.x * blockDim.x + threadIdx.x; i < total; i += stride) {
    int n = i / K, k = i - n * K;
    out[i] = (_Float16)in[(size_t)(rowOff + k) * N + n];
  }
}

// ---------------------------------------------------------------------------
// Encoder GRU scan: single persistent workgroup (16 waves), h16 lives in LDS.
// Per step: gh = h16 @ WhT (96 WMMA tiles), then Keras-GRU gates using the
// precomputed GX = emb@Wx + b, emitting enc_out (f32 + f16).
// Dynamic LDS: h16 16KB | gh 96KB
// ---------------------------------------------------------------------------
__global__ __launch_bounds__(512) void enc_scan(
    const _Float16* __restrict__ WhT, const float* __restrict__ GX,
    float* __restrict__ enc_out, _Float16* __restrict__ enc_out16,
    _Float16* __restrict__ h16_out) {
  extern __shared__ char smem[];
  _Float16* h16 = (_Float16*)smem;                   // [32][256]
  float*    gh  = (float*)(smem + 16384);            // [32][768]
  int tid = threadIdx.x, wv = tid >> 5;

  for (int i = tid; i < BB * UU; i += 512) h16[i] = (_Float16)0.f;
  __syncthreads();

  for (int t = 0; t < SS; ++t) {
    // gh = h @ Wh  (M=32 -> 2 mtiles, N=768 -> 48 ntiles)
    for (int j = wv; j < 96; j += 16) {
      int mi = j & 1, ni = j >> 1;
      v8f acc = {};
      acc = wmma_strip(h16 + mi * 16 * UU, UU, WhT + ni * 16 * UU, UU, UU, acc);
      store_tile(gh + mi * 16 * G3 + ni * 16, G3, acc, nullptr);
    }
    __syncthreads();
    // gates
    for (int i = tid; i < BB * UU; i += 512) {
      int b = i >> 8, u = i & 255;
      const float* gx = GX + (size_t)(b * SS + t) * G3;
      float z  = sigf(gx[u]        + gh[b * G3 + u]);
      float r  = sigf(gx[256 + u]  + gh[b * G3 + 256 + u]);
      float hh = tanhf(gx[512 + u] + r * gh[b * G3 + 512 + u]);
      float hn = z * (float)h16[i] + (1.f - z) * hh;
      enc_out  [(size_t)(b * SS + t) * UU + u] = hn;
      enc_out16[(size_t)(b * SS + t) * UU + u] = (_Float16)hn;
      h16[i] = (_Float16)hn;
    }
    __syncthreads();
  }
  for (int i = tid; i < BB * UU; i += 512) h16_out[i] = h16[i];
}

// ---------------------------------------------------------------------------
// One decoder step (Bahdanau attention + GRU), single workgroup.
// Dynamic LDS: h16 16K | ctx16 16K | W2h 32K | sc 16K | gx 96K | gh 96K = 272K
// ---------------------------------------------------------------------------
__global__ __launch_bounds__(512) void dec_step(
    const _Float16* __restrict__ W2T, const float* __restrict__ b2,
    const float* __restrict__ Va, const float* __restrict__ ba,
    const float* __restrict__ enc_proj, const float* __restrict__ enc_out,
    const _Float16* __restrict__ WxTopT, const float* __restrict__ GXE,
    const _Float16* __restrict__ WhT, _Float16* __restrict__ h16_g, int t) {
  extern __shared__ char smem[];
  _Float16* h16   = (_Float16*)smem;                 // [32][256]
  _Float16* ctx16 = (_Float16*)(smem + 16384);       // [32][256]
  float*    W2h   = (float*)(smem + 32768);          // [32][256]
  float*    sc    = (float*)(smem + 65536);          // [32][128]
  float*    gx    = (float*)(smem + 81920);          // [32][768]
  float*    gh    = (float*)(smem + 180224);         // [32][768]
  int tid = threadIdx.x, wv = tid >> 5, lane = tid & 31;

  for (int i = tid; i < BB * UU; i += 512) h16[i] = h16_g[i];
  __syncthreads();

  // W2h = h @ W2 + b2 (2 x 16 tiles)
  for (int j = wv; j < 32; j += 16) {
    int mi = j & 1, ni = j >> 1;
    v8f acc = {};
    acc = wmma_strip(h16 + mi * 16 * UU, UU, W2T + ni * 16 * UU, UU, UU, acc);
    store_tile(W2h + mi * 16 * UU + ni * 16, UU, acc, b2 + ni * 16);
  }
  __syncthreads();

  // scores[b,s] = tanh(enc_proj[b,s,:] + W2h[b,:]) . Va + ba
  for (int p = tid; p < BB * SS; p += 512) {
    int b = p >> 7, s = p & 127;
    const float* ep = enc_proj + (size_t)(b * SS + s) * UU;
    float acc = 0.f;
    for (int u = 0; u < UU; ++u) acc += tanhf(ep[u] + W2h[b * UU + u]) * Va[u];
    sc[p] = acc + ba[0];
  }
  __syncthreads();

  // softmax over s per batch row (one wave per row, 2 rows/wave)
  for (int rr = 0; rr < 2; ++rr) {
    int b = wv + rr * 16;
    float* row = sc + b * SS;
    float m = -3.0e38f;
    for (int s = lane; s < SS; s += 32) m = fmaxf(m, row[s]);
    for (int o = 16; o; o >>= 1) m = fmaxf(m, __shfl_xor(m, o, 32));
    float sum = 0.f;
    for (int s = lane; s < SS; s += 32) { float e = __expf(row[s] - m); row[s] = e; sum += e; }
    for (int o = 16; o; o >>= 1) sum += __shfl_xor(sum, o, 32);
    float inv = 1.f / sum;
    for (int s = lane; s < SS; s += 32) row[s] *= inv;
  }
  __syncthreads();

  // ctx[b,u] = sum_s w[b,s] * enc_out[b,s,u]
  for (int i = tid; i < BB * UU; i += 512) {
    int b = i >> 8, u = i & 255;
    const float* eo = enc_out + (size_t)b * SS * UU + u;
    const float* wr = sc + b * SS;
    float acc = 0.f;
    for (int s = 0; s < SS; ++s) acc += wr[s] * eo[s * UU];
    ctx16[i] = (_Float16)acc;
  }
  __syncthreads();

  // gh = h @ dec_Wh ; gx = ctx @ dec_Wx_top + GXE[b,t,:]  (192 tiles total)
  for (int j = wv; j < 192; j += 16) {
    if (j < 96) {
      int mi = j & 1, ni = j >> 1;
      v8f acc = {};
      acc = wmma_strip(h16 + mi * 16 * UU, UU, WhT + ni * 16 * UU, UU, UU, acc);
      store_tile(gh + mi * 16 * G3 + ni * 16, G3, acc, nullptr);
    } else {
      int jj = j - 96;
      int mi = jj & 1, ni = jj >> 1;
      v8f acc = {};
      acc = wmma_strip(ctx16 + mi * 16 * UU, UU, WxTopT + ni * 16 * UU, UU, UU, acc);
      int g = lane >> 4, n = lane & 15;
#pragma unroll
      for (int r = 0; r < 8; ++r) {
        int m = mi * 16 + r + 8 * g;
        int col = ni * 16 + n;
        gx[m * G3 + col] = acc[r] + GXE[(size_t)(m * SS + t) * G3 + col];
      }
    }
  }
  __syncthreads();

  // GRU gates -> h_new
  for (int i = tid; i < BB * UU; i += 512) {
    int b = i >> 8, u = i & 255;
    float z  = sigf(gx[b * G3 + u]        + gh[b * G3 + u]);
    float r  = sigf(gx[b * G3 + 256 + u]  + gh[b * G3 + 256 + u]);
    float hh = tanhf(gx[b * G3 + 512 + u] + r * gh[b * G3 + 512 + u]);
    float hn = z * (float)h16[i] + (1.f - z) * hh;
    h16_g[i] = (_Float16)hn;
  }
}

// ---------------------------------------------------------------------------
// Host side
// ---------------------------------------------------------------------------
extern "C" void kernel_launch(void* const* d_in, const int* in_sizes, int n_in,
                              void* d_out, int out_size, void* d_ws, size_t ws_size,
                              hipStream_t stream) {
  (void)in_sizes; (void)n_in; (void)out_size; (void)ws_size;
  const int*   x         = (const int*)  d_in[0];
  const int*   labels    = (const int*)  d_in[1];
  const float* enc_embed = (const float*)d_in[2];
  const float* enc_Wx    = (const float*)d_in[3];
  const float* enc_Wh    = (const float*)d_in[4];
  const float* enc_b     = (const float*)d_in[5];
  const float* dec_embed = (const float*)d_in[6];
  const float* dec_Wx    = (const float*)d_in[7];
  const float* dec_Wh    = (const float*)d_in[8];
  const float* dec_b     = (const float*)d_in[9];
  const float* W1        = (const float*)d_in[10];
  const float* b1        = (const float*)d_in[11];
  const float* W2        = (const float*)d_in[12];
  const float* b2        = (const float*)d_in[13];
  const float* Va        = (const float*)d_in[14];
  const float* ba        = (const float*)d_in[15];
  const float* Wfc       = (const float*)d_in[16];
  const float* bfc       = (const float*)d_in[17];
  float* out = (float*)d_out;

  // workspace carve (256B aligned)
  char* ws = (char*)d_ws;
  size_t off = 0;
  auto carve = [&](size_t bytes) -> void* {
    void* p = ws + off;
    off += (bytes + 255) & ~((size_t)255);
    return p;
  };
  const int BS = BB * SS;                       // 4096
  _Float16* encA16     = (_Float16*)carve((size_t)BS * EE * 2);
  _Float16* decE16     = (_Float16*)carve((size_t)BS * EE * 2);
  _Float16* enc_WxT    = (_Float16*)carve((size_t)G3 * EE * 2);
  _Float16* enc_WhT    = (_Float16*)carve((size_t)G3 * UU * 2);
  _Float16* W1T        = (_Float16*)carve((size_t)UU * UU * 2);
  _Float16* W2T        = (_Float16*)carve((size_t)UU * UU * 2);
  _Float16* WxTopT     = (_Float16*)carve((size_t)G3 * UU * 2);
  _Float16* WxBotT     = (_Float16*)carve((size_t)G3 * EE * 2);
  _Float16* dec_WhT    = (_Float16*)carve((size_t)G3 * UU * 2);
  _Float16* WfcT       = (_Float16*)carve((size_t)VV * UU * 2);
  float*    GXenc      = (float*)carve((size_t)BS * G3 * 4);
  float*    GXE        = (float*)carve((size_t)BS * G3 * 4);
  float*    enc_out    = (float*)carve((size_t)BS * UU * 4);
  _Float16* enc_out16  = (_Float16*)carve((size_t)BS * UU * 2);
  float*    enc_proj   = (float*)carve((size_t)BS * UU * 4);
  _Float16* h16        = (_Float16*)carve((size_t)BB * UU * 2);

  auto gemm = [&](const _Float16* A, const _Float16* Bt, const float* bias,
                  float* C, int ldc, int M, int N, int K) {
    int jobs = (M / 32) * (N / 32);      // 32x32 block per wave
    int blocks = (jobs + 7) / 8;
    gemm_f16_wmma<<<blocks, 256, 0, stream>>>(A, Bt, bias, C, ldc, M, N, K);
  };

  // 1) embedding gathers -> f16 activation matrices
  gather_embed_f16<<<1024, 256, 0, stream>>>(x, x, enc_embed, encA16, BS, EE, SS, 0);
  gather_embed_f16<<<1024, 256, 0, stream>>>(x, labels, dec_embed, decE16, BS, EE, SS, 1);

  // 2) weight transposed-f16 conversions
  convT_f16<<<512,  256, 0, stream>>>(enc_Wx, enc_WxT, G3, EE, 0);
  convT_f16<<<512,  256, 0, stream>>>(enc_Wh, enc_WhT, G3, UU, 0);
  convT_f16<<<256,  256, 0, stream>>>(W1,     W1T,     UU, UU, 0);
  convT_f16<<<256,  256, 0, stream>>>(W2,     W2T,     UU, UU, 0);
  convT_f16<<<512,  256, 0, stream>>>(dec_Wx, WxTopT,  G3, UU, 0);
  convT_f16<<<512,  256, 0, stream>>>(dec_Wx, WxBotT,  G3, EE, UU);
  convT_f16<<<512,  256, 0, stream>>>(dec_Wh, dec_WhT, G3, UU, 0);
  convT_f16<<<4096, 256, 0, stream>>>(Wfc,    WfcT,    VV, UU, 0);

  // 3) batched time-parallel GEMMs
  gemm(encA16, enc_WxT, enc_b, GXenc, G3, BS, G3, EE);   // emb@enc_Wx + b
  gemm(decE16, WxBotT,  dec_b, GXE,   G3, BS, G3, EE);   // emb@dec_Wx[bot] + b

  // 4) encoder GRU scan (persistent single WG, LDS-resident h)
  enc_scan<<<1, 512, 16384 + BB * G3 * 4, stream>>>(enc_WhT, GXenc, enc_out,
                                                    enc_out16, h16);

  // 5) enc_proj = enc_out @ W1 + b1
  gemm(enc_out16, W1T, b1, enc_proj, UU, BS, UU, UU);

  // 6) decoder: 128 sequential steps; each step = attention+GRU (1 WG)
  //    followed by the full-GPU logits WMMA GEMM into d_out[B,S,V].
  const size_t decSmem = 16384 + 16384 + 32768 + 16384 + 98304 + 98304;
  for (int t = 0; t < SS; ++t) {
    dec_step<<<1, 512, decSmem, stream>>>(W2T, b2, Va, ba, enc_proj, enc_out,
                                          WxTopT, GXE, dec_WhT, h16, t);
    // logits[b, t, v]: C = out + t*V, ldc = S*V  ->  out[(b*S+t)*V + v]
    gemm(h16, WfcT, bfc, out + (size_t)t * VV, SS * VV, BB, VV, UU);
  }
}
